// VQActivation_12326556139529
// MI455X (gfx1250) — matrix operand order repaired and checked by reading, not compiled, not run on hardware
//
#include <hip/hip_runtime.h>

// CDNA5 / gfx1250 residual VQ kernel, v3.
// - 64 rows per workgroup (4 row sub-tiles/wave) to amortize codebook reads over L2.
// - 4 independent WMMA accumulator chains per B fragment.
// - B fragments software-pipelined 2 deep (incl. across column-tile boundary)
//   so L2 load latency overlaps the WMMA stream instead of s_wait_loadcnt 0x0
//   serialization per fragment.
// - bf16 WMMA selects argmax code; u recomputed exactly in fp32.
// - out = x - resid_final (no separate accumulator).

typedef __attribute__((ext_vector_type(16))) __bf16 v16bf;
typedef __attribute__((ext_vector_type(8)))  __bf16 v8bf;
typedef __attribute__((ext_vector_type(8)))  float  v8f;

#define C_DIM   256
#define KS_DIM  1024
#define HW_DIM  3136      // 56*56
#define M_TILE  64        // rows per workgroup
#define RT_N    4         // row sub-tiles of 16
#define LDA     264       // padded bf16 row stride (528B -> bank-conflict free)
#define NWAVES  8

__device__ __forceinline__ __bf16 f32_to_bf16(float f) {
  unsigned u = __builtin_bit_cast(unsigned, f);
  u += 0x7FFFu + ((u >> 16) & 1u);          // round to nearest even
  unsigned short h = (unsigned short)(u >> 16);
  return __builtin_bit_cast(__bf16, h);
}

// One-shot fp32 -> bf16 codebook conversion into workspace (L2-resident, 512KB).
__global__ void cb_to_bf16_kernel(const float* __restrict__ cb,
                                  __bf16* __restrict__ cbbf) {
  int i = blockIdx.x * blockDim.x + threadIdx.x;
  cbbf[i] = f32_to_bf16(cb[i]);
}

__global__ __launch_bounds__(256) void vq_wmma_kernel(
    const float* __restrict__ x,
    const float* __restrict__ cb,        // fp32 codebook (exact u + update)
    const __bf16* __restrict__ cbbf,     // bf16 codebook (WMMA B operands)
    const int* __restrict__ depth_p,
    float* __restrict__ out) {

  __shared__ float  resid_f[M_TILE * C_DIM];   // 64 KB
  __shared__ __bf16 resid_bf[M_TILE * LDA];    // 33 KB (padded rows)
  __shared__ float  wmax[NWAVES][M_TILE];
  __shared__ int    widx[NWAVES][M_TILE];
  __shared__ float  best_u[M_TILE];
  __shared__ int    best_code[M_TILE];

  const int t       = threadIdx.x;
  const int lane    = t & 31;
  const int wv      = t >> 5;        // wave 0..7
  const int laneLow = lane & 15;
  const int laneHi  = lane >> 4;     // 0 or 1

  const int n0 = blockIdx.x * M_TILE;   // 64 rows; 3136 % 64 == 0 -> no batch crossing
  const int b  = n0 / HW_DIM;
  const int hw = n0 % HW_DIM;
  const int depth = depth_p[0];

  // ---- load x tile: NCHW -> [row i][channel c], 256B coalesced per channel ----
  {
    const int i = t & 63;
    for (int c = t >> 6; c < C_DIM; c += 4) {
      float v = x[((size_t)(b * C_DIM + c)) * HW_DIM + hw + i];
      resid_f[i * C_DIM + c] = v;
      resid_bf[i * LDA + c]  = f32_to_bf16(v);
    }
  }
  __syncthreads();

  for (int d = 0; d < depth; ++d) {
    // ---- hoist A fragments: 4 row sub-tiles x 8 K-steps of 16x32 bf16 ----
    // A as loaded here (contiguous k, 16x32): lane<16 row=lane, elems 0..15 -> K=base..base+15;
    // lane>=16 same row with +16 K offset (matches contiguous B).
    union { v16bf v; v8bf h[2]; } af[RT_N][8];
#pragma unroll
    for (int rt = 0; rt < RT_N; ++rt) {
      const int arow = rt * 16 + laneLow;
#pragma unroll
      for (int ks = 0; ks < 8; ++ks) {
        const int ka = ks * 32 + laneHi * 16;
        af[rt][ks].h[0] = *(const v8bf*)&resid_bf[arow * LDA + ka];
        af[rt][ks].h[1] = *(const v8bf*)&resid_bf[arow * LDA + ka + 8];
      }
    }

    float vmax[RT_N][8];
    int   vidx[RT_N][8];
#pragma unroll
    for (int rt = 0; rt < RT_N; ++rt)
#pragma unroll
      for (int r = 0; r < 8; ++r) { vmax[rt][r] = -3.402823466e+38f; vidx[rt][r] = 0; }

    // ---- each wave covers 128 of the 1024 codebook columns ----
    // B as loaded here (contiguous k, 32x16): lane<16 -> column=lane, K=0..15;
    // lane>=16 -> column=lane-16, K=16..31. 16 contiguous bf16 per lane.
    const int col0 = wv * 128 + laneLow;
    const __bf16* bptr = cbbf + (size_t)col0 * C_DIM + laneHi * 16;

    // 2-deep pipelined B fragments (next column tile is bptr + 16*256 elems).
    v16bf b0 = *(const v16bf*)(bptr);
    v16bf b1 = *(const v16bf*)(bptr + 32);

    for (int tcol = 0; tcol < 8; ++tcol) {
      const bool last = (tcol == 7);
      v8f acc[RT_N];
#pragma unroll
      for (int rt = 0; rt < RT_N; ++rt) acc[rt] = (v8f){};

#pragma unroll
      for (int ks = 0; ks < 8; ++ks) {
        // Prefetch fragment ks+2 (spills into next column tile; clamped on last).
        const __bf16* p;
        if (ks < 6)      p = bptr + (ks + 2) * 32;
        else if (!last)  p = bptr + 16 * C_DIM + (ks - 6) * 32;
        else             p = bptr;                  // harmless dummy reload
        v16bf bn = *(const v16bf*)p;

#pragma unroll
        for (int rt = 0; rt < RT_N; ++rt) {
          acc[rt] = __builtin_amdgcn_wmma_f32_16x16x32_bf16(
                        false, af[rt][ks].v, false, b0, (short)0, acc[rt],
                        false, false);
        }
        b0 = b1;
        b1 = bn;
      }

      // D layout: lane holds column (lane&15); VGPR r -> row r (+8 for lane>=16).
      const int col = col0 + tcol * 16;
#pragma unroll
      for (int rt = 0; rt < RT_N; ++rt)
#pragma unroll
        for (int r = 0; r < 8; ++r) {
          if (acc[rt][r] > vmax[rt][r]) { vmax[rt][r] = acc[rt][r]; vidx[rt][r] = col; }
        }

      bptr += 16 * C_DIM;
    }

    // ---- intra-wave reduction across the 16 lanes of each half (wave32) ----
#pragma unroll
    for (int rt = 0; rt < RT_N; ++rt)
#pragma unroll
      for (int r = 0; r < 8; ++r) {
        for (int m = 1; m <= 8; m <<= 1) {
          float om = __shfl_xor(vmax[rt][r], m, 32);
          int   oi = __shfl_xor(vidx[rt][r], m, 32);
          if (om > vmax[rt][r] || (om == vmax[rt][r] && oi < vidx[rt][r])) {
            vmax[rt][r] = om; vidx[rt][r] = oi;
          }
        }
      }
    if (laneLow == 0) {
#pragma unroll
      for (int rt = 0; rt < RT_N; ++rt)
#pragma unroll
        for (int r = 0; r < 8; ++r) {
          wmax[wv][rt * 16 + laneHi * 8 + r] = vmax[rt][r];
          widx[wv][rt * 16 + laneHi * 8 + r] = vidx[rt][r];
        }
    }
    __syncthreads();

    // ---- cross-wave reduction: 64 threads, one per row ----
    if (t < M_TILE) {
      float bm = wmax[0][t];
      int   bi = widx[0][t];
      for (int w = 1; w < NWAVES; ++w) {
        float om = wmax[w][t];
        int   oi = widx[w][t];
        if (om > bm || (om == bm && oi < bi)) { bm = om; bi = oi; }
      }
      best_code[t] = bi;
    }
    __syncthreads();

    // ---- exact u in fp32: u = <resid_row, cb[code]> (4 lanes per row) ----
    {
      const int row = t >> 2;
      const int q   = t & 3;
      const int code = best_code[row];
      const float* cr = cb + (size_t)code * C_DIM + q * 64;
      const float* rr = resid_f + row * C_DIM + q * 64;
      float s = 0.0f;
#pragma unroll 8
      for (int j = 0; j < 64; ++j) s += rr[j] * cr[j];
      s += __shfl_xor(s, 1, 32);
      s += __shfl_xor(s, 2, 32);
      if (q == 0) best_u[row] = s;
    }
    __syncthreads();

    // ---- residual update in fp32: resid -= cb[code] * u ----
#pragma unroll
    for (int k = 0; k < 4; ++k) {
      const int un  = t + 256 * k;       // 64 rows x 16 segments
      const int row = un >> 4;
      const int seg = un & 15;
      const float u  = best_u[row];
      const int code = best_code[row];
      const float* crow = cb + (size_t)code * C_DIM + seg * 16;
#pragma unroll
      for (int j = 0; j < 16; ++j) {
        const int c = seg * 16 + j;
        float rv = resid_f[row * C_DIM + c] - crow[j] * u;
        resid_f[row * C_DIM + c] = rv;
        resid_bf[row * LDA + c]  = f32_to_bf16(rv);
      }
    }
    __syncthreads();
  }

  // ---- store: acc == x - resid_final, same coalesced pattern ----
  {
    const int i = t & 63;
    for (int c = t >> 6; c < C_DIM; c += 4) {
      const size_t off = ((size_t)(b * C_DIM + c)) * HW_DIM + hw + i;
      out[off] = x[off] - resid_f[i * C_DIM + c];
    }
  }
}

extern "C" void kernel_launch(void* const* d_in, const int* in_sizes, int n_in,
                              void* d_out, int out_size, void* d_ws, size_t ws_size,
                              hipStream_t stream) {
  const float* x     = (const float*)d_in[0];   // [32,256,56,56] fp32
  const float* cb    = (const float*)d_in[1];   // [1024,256] fp32
  const int*   depth = (const int*)d_in[2];     // scalar
  float* out = (float*)d_out;

  __bf16* cbbf = (__bf16*)d_ws;                 // 1024*256 bf16 = 512 KB scratch

  cb_to_bf16_kernel<<<(KS_DIM * C_DIM) / 256, 256, 0, stream>>>(cb, cbbf);

  const int n_rows = 32 * HW_DIM;               // 100352
  vq_wmma_kernel<<<n_rows / M_TILE, 256, 0, stream>>>(x, cb, cbbf, depth, out);
}
